// SequenceEncoder_36627481101152
// MI455X (gfx1250) — compile-verified, matrix operand order checked by reading, not run on hardware
//
#include <hip/hip_runtime.h>
#include <hip/hip_bf16.h>

// ---------------------------------------------------------------------------
// DIN-style sequence encoder for MI455X (gfx1250), wave32 + bf16 WMMA.
//
// Decomposition: [q,k,q-k,q*k]@W1 == kp@W1eff_b + bias_b  with
//   W1eff_b = (W1b - W1c) + diag(q_b) * W1d ,  bias_b = b1 + q_b @ (W1a + W1c)
// so layer-1 K shrinks 512 -> 128 and the WMMA A operand is just kp (bf16).
//
// This revision software-pipelines the A-fragment ds_load_b128 traffic:
// distinct registers per fragment + cross-tile prefetch of the (read-only)
// kp16 fragments so WMMA issue is not serialized on s_wait_dscnt.
// ---------------------------------------------------------------------------

typedef __attribute__((ext_vector_type(16))) __bf16 v16bf;
typedef __attribute__((ext_vector_type(8)))  __bf16 v8bf;
typedef __attribute__((ext_vector_type(4)))  __bf16 v4bf;
typedef __attribute__((ext_vector_type(8)))  float  v8f;

#define LQ    200            // sequence length
#define LPAD  208            // padded to 13 * 16
#define EDIM  128            // embedding dim
#define ADIM  128            // attention MLP dim
#define MT    13             // 16-row M tiles
#define KPS   136            // padded row stride (bf16 elems): 272B -> bank spread

// dynamic LDS carve (bytes, all 16B aligned)
#define KP16_OFF  0                       // 208*136*2 = 56576
#define W1E_OFF   56576                   // 128*136*2 = 34816
#define H1_OFF    91392                   // 16*136*2  = 4352
#define QS_OFF    95744                   // 128*4     = 512
#define BIAS_OFF  96256                   // 128*4     = 512
#define SC_OFF    96768                   // 208*4     = 832
#define RED_OFF   97600                   // 256*4     = 1024
#define SMEM_BYTES 98624

// ---------------------------------------------------------------------------
// Prep: fold W1 blocks, transpose for coalesced per-block W1eff build, and
// pre-swizzle W2 into the wave32 WMMA B-fragment layout (16-bit, 16x16x32):
//   lane<16  : K = kk*32 + {0..7, 16..23},  col = lane
//   lane>=16 : K = kk*32 + {8..15, 24..31}, col = lane-16
// ---------------------------------------------------------------------------
__global__ void __launch_bounds__(256) din_prep(
    const float* __restrict__ W1, const float* __restrict__ W2,
    float* __restrict__ W1q, float* __restrict__ W1kT, float* __restrict__ W1pT,
    __bf16* __restrict__ w2f) {
  int tid = blockIdx.x * blockDim.x + threadIdx.x;
  if (tid < EDIM * ADIM) {
    int e = tid >> 7, n = tid & 127;
    float a  = W1[(0   + e) * ADIM + n];
    float bb = W1[(128 + e) * ADIM + n];
    float c  = W1[(256 + e) * ADIM + n];
    float d  = W1[(384 + e) * ADIM + n];
    W1q [e * ADIM + n] = a + c;       // [e][n] : coalesced per-n reads later
    W1kT[n * EDIM + e] = bb - c;      // [n][e] : coalesced W1eff build
    W1pT[n * EDIM + e] = d;
  } else if (tid < 2 * EDIM * ADIM) {
    int t    = tid - EDIM * ADIM;     // W2 fragment swizzle
    int j    = t & 15;
    int lane = (t >> 4) & 31;
    int kk   = (t >> 9) & 3;
    int nt   = t >> 11;
    int n    = nt * 16 + (lane & 15);
    int half = lane >> 4;
    int k    = kk * 32 + ((j < 8) ? (half * 8 + j) : (16 + half * 8 + (j - 8)));
    w2f[t] = (__bf16)W2[k * ADIM + n];
  }
}

__device__ __forceinline__ void load_a_frag(const __bf16* base, v16bf& a) {
  v8bf lo = *(const v8bf*)base;          // ds_load_b128 (16B)
  v8bf hi = *(const v8bf*)(base + 16);   // ds_load_b128 (+32B)
#pragma unroll
  for (int j = 0; j < 8; ++j) { a[j] = lo[j]; a[8 + j] = hi[j]; }
}

// ---------------------------------------------------------------------------
// Main: one workgroup (8 wave32) per batch row.
// ---------------------------------------------------------------------------
__global__ void __launch_bounds__(256) din_main(
    const float* __restrict__ query, const float* __restrict__ keys,
    const long long* __restrict__ klen, const float* __restrict__ pos,
    const float* __restrict__ b1, const float* __restrict__ b2,
    const float* __restrict__ W3, const float* __restrict__ b3,
    const float* __restrict__ W1q, const float* __restrict__ W1kT,
    const float* __restrict__ W1pT, const __bf16* __restrict__ w2f,
    float* __restrict__ out) {
  extern __shared__ char smem[];
  __bf16* kp16   = (__bf16*)(smem + KP16_OFF);   // [208][136] keys + pos_emb
  __bf16* w1effT = (__bf16*)(smem + W1E_OFF);    // [128][136] per-batch W1eff^T
  __bf16* h1T    = (__bf16*)(smem + H1_OFF);     // [16][136]  layer-1 tile
  float*  qs     = (float*)(smem + QS_OFF);
  float*  bias1  = (float*)(smem + BIAS_OFF);
  float*  sc     = (float*)(smem + SC_OFF);      // scores (atomic accum)
  float*  red    = (float*)(smem + RED_OFF);

  const int b    = blockIdx.x;
  const int tid  = threadIdx.x;
  const int lane = tid & 31;
  const int wv   = tid >> 5;          // wave id = N tile id (0..7)
  const int rw   = lane & 15;         // row-in-tile (A) / col-in-tile (B/C/D)
  const int hf   = lane >> 4;         // K-half selector

  if (tid < EDIM) qs[tid] = query[(size_t)b * EDIM + tid];
  if (tid < LPAD) sc[tid] = 0.0f;
  __syncthreads();

  // kp = keys + pos_emb -> bf16, padded row stride; pad rows zeroed
  {
    const float4* k4 = (const float4*)(keys + (size_t)b * LQ * EDIM);
    const float4* p4 = (const float4*)pos;
    for (int i = tid; i < LQ * EDIM / 4; i += 256) {
      float4 kv = k4[i], pv = p4[i];
      int r = i >> 5, c = (i & 31) * 4;
      v4bf v;
      v[0] = (__bf16)(kv.x + pv.x); v[1] = (__bf16)(kv.y + pv.y);
      v[2] = (__bf16)(kv.z + pv.z); v[3] = (__bf16)(kv.w + pv.w);
      *(v4bf*)(kp16 + r * KPS + c) = v;
    }
    for (int i = tid; i < (LPAD - LQ) * EDIM; i += 256) {
      int r = LQ + (i >> 7), c = i & 127;
      kp16[r * KPS + c] = (__bf16)0.0f;
    }
  }

  // per-batch folded bias partials: all 256 threads, half the e-range each
  {
    int n = tid & 127, h2 = tid >> 7;
    float acc = 0.0f;
    for (int e = h2 * 64; e < h2 * 64 + 64; ++e) acc += qs[e] * W1q[e * ADIM + n];
    red[tid] = acc;
  }
  // per-batch W1eff^T[n][e] = (W1b-W1c)[e][n] + q[e]*W1d[e][n]  (bf16)
  for (int i = tid; i < ADIM * EDIM; i += 256) {
    int n = i >> 7, e = i & 127;
    float v = W1kT[i] + qs[e] * W1pT[i];
    w1effT[n * KPS + e] = (__bf16)v;
  }
  __syncthreads();
  if (tid < ADIM) bias1[tid] = b1[tid] + red[tid] + red[tid + 128];
  __syncthreads();

  // register-resident B fragments for both layers (this wave's 16 columns)
  const int col = wv * 16 + rw;
  v16bf B1f[4], B2f[4];
#pragma unroll
  for (int kk = 0; kk < 4; ++kk)
    load_a_frag(w1effT + col * KPS + kk * 32 + hf * 8, B1f[kk]);
  {
    const v16bf* p2 = (const v16bf*)w2f;
#pragma unroll
    for (int kk = 0; kk < 4; ++kk) B2f[kk] = p2[(wv * 4 + kk) * 32 + lane];
  }
  const float bias1v = bias1[col];
  const float b2v    = b2[col];
  const float w3v    = W3[col];

  // preload tile 0 layer-1 A fragments (kp16 is read-only from here on)
  const __bf16* aBase = kp16 + rw * KPS + hf * 8;
  v16bf a1[4];
#pragma unroll
  for (int kk = 0; kk < 4; ++kk) load_a_frag(aBase + kk * 32, a1[kk]);

  // main loop over 16-row M tiles (software-pipelined A fragments)
  for (int mt = 0; mt < MT; ++mt) {
    // ---- layer 1: h1 = relu(kp @ W1eff + bias1) ----
    v8f acc = {};
#pragma unroll
    for (int kk = 0; kk < 4; ++kk)
      acc = __builtin_amdgcn_wmma_f32_16x16x32_bf16(
          false, a1[kk], false, B1f[kk], (short)0, acc, false, false);

    // prefetch next tile's layer-1 fragments; latency hides under the
    // relu/convert/store + barrier + layer-2 phase below
    v16bf a1n[4];
    if (mt + 1 < MT) {
      const __bf16* nb = aBase + (mt + 1) * 16 * KPS;
#pragma unroll
      for (int kk = 0; kk < 4; ++kk) load_a_frag(nb + kk * 32, a1n[kk]);
    }

#pragma unroll
    for (int i = 0; i < 8; ++i) {
      float h = acc[i] + bias1v;
      h = h > 0.0f ? h : 0.0f;
      h1T[(i + hf * 8) * KPS + col] = (__bf16)h;   // C/D layout -> row-major
    }
    __syncthreads();

    // ---- layer 2: h2 = relu(h1 @ W2 + b2); scores += h2 @ W3 ----
    v16bf a2[4];
#pragma unroll
    for (int kk = 0; kk < 4; ++kk)
      load_a_frag(h1T + rw * KPS + kk * 32 + hf * 8, a2[kk]);
    v8f acc2 = {};
#pragma unroll
    for (int kk = 0; kk < 4; ++kk)
      acc2 = __builtin_amdgcn_wmma_f32_16x16x32_bf16(
          false, a2[kk], false, B2f[kk], (short)0, acc2, false, false);

#pragma unroll
    for (int i = 0; i < 8; ++i) {
      float h = acc2[i] + b2v;
      h = h > 0.0f ? h : 0.0f;
      float v = h * w3v;                 // partial score, this wave's 16 cols
      v += __shfl_xor(v, 1, 32);         // reduce across 16-lane half
      v += __shfl_xor(v, 2, 32);
      v += __shfl_xor(v, 4, 32);
      v += __shfl_xor(v, 8, 32);
      if ((lane & 15) == 0) atomicAdd(&sc[mt * 16 + i + hf * 8], v);
    }
    __syncthreads();

    if (mt + 1 < MT) {
#pragma unroll
      for (int kk = 0; kk < 4; ++kk) a1[kk] = a1n[kk];
    }
  }

  // ---- masked softmax over L ----
  const long long len = klen[b];
  const float b3v = b3[0];
  float sval = -3.0e38f;
  if (tid < LQ) {
    float s = sc[tid] + b3v;
    if ((long long)tid >= len) s = -1.0e9f;
    sval = s;
  }
  red[tid] = sval;
  __syncthreads();
#pragma unroll
  for (int st = 128; st >= 1; st >>= 1) {
    if (tid < st) { float o = red[tid + st]; if (o > red[tid]) red[tid] = o; }
    __syncthreads();
  }
  const float mx = red[0];
  __syncthreads();
  float ex = (tid < LQ) ? __expf(sval - mx) : 0.0f;
  if (tid < LPAD) sc[tid] = ex;
  red[tid] = ex;
  __syncthreads();
#pragma unroll
  for (int st = 128; st >= 1; st >>= 1) {
    if (tid < st) red[tid] += red[tid + st];
    __syncthreads();
  }
  const float inv = 1.0f / red[0];
  __syncthreads();   // red[0] consumed by all before red is reused below

  // ---- out[b] = softmax(scores) @ kp : 256 threads, half-L each ----
  {
    int e = tid & 127, h2 = tid >> 7;
    float acc = 0.0f;
    for (int l = h2 * 100; l < h2 * 100 + 100; ++l)
      acc += sc[l] * (float)kp16[l * KPS + e];
    red[tid] = acc;
  }
  __syncthreads();
  if (tid < EDIM)
    out[(size_t)b * EDIM + tid] = (red[tid] + red[tid + 128]) * inv;
}

extern "C" void kernel_launch(void* const* d_in, const int* in_sizes, int n_in,
                              void* d_out, int out_size, void* d_ws, size_t ws_size,
                              hipStream_t stream) {
  const float*     query = (const float*)d_in[0];
  const float*     keys  = (const float*)d_in[1];
  const long long* klen  = (const long long*)d_in[2];   // int64 in reference
  const float*     pos   = (const float*)d_in[3];
  const float*     W1    = (const float*)d_in[4];
  const float*     b1    = (const float*)d_in[5];
  const float*     W2    = (const float*)d_in[6];
  const float*     b2    = (const float*)d_in[7];
  const float*     W3    = (const float*)d_in[8];
  const float*     b3    = (const float*)d_in[9];

  const int Bn = in_sizes[0] / EDIM;                    // 4096

  float*  W1q  = (float*)d_ws;                          // [128][128]
  float*  W1kT = W1q  + EDIM * ADIM;                    // [128][128]
  float*  W1pT = W1kT + EDIM * ADIM;                    // [128][128]
  __bf16* w2f  = (__bf16*)(W1pT + EDIM * ADIM);         // 16384 bf16 frags

  (void)hipFuncSetAttribute((const void*)din_main,
                            hipFuncAttributeMaxDynamicSharedMemorySize,
                            SMEM_BYTES);

  din_prep<<<(2 * EDIM * ADIM + 255) / 256, 256, 0, stream>>>(
      W1, W2, W1q, W1kT, W1pT, w2f);
  din_main<<<Bn, 256, SMEM_BYTES, stream>>>(
      query, keys, klen, pos, b1, b2, W3, b3, W1q, W1kT, W1pT, w2f,
      (float*)d_out);
}
